// mfb_coatt_embed_ocr_binhelp_55851754717187
// MI455X (gfx1250) — matrix operand order repaired and checked by reading, not compiled, not run on hardware
//
#include <hip/hip_runtime.h>

// ---------------------------------------------------------------------------
// Types for CDNA5 WMMA (wave32, 16x16x32 bf16 -> f32)
// ---------------------------------------------------------------------------
typedef __bf16 bf16;
typedef __attribute__((ext_vector_type(8)))  __bf16 v8bf;
typedef __attribute__((ext_vector_type(16))) __bf16 v16bf;
typedef __attribute__((ext_vector_type(8)))  float  v8f;

#define GSL(i, n) for (long i = (long)blockIdx.x * blockDim.x + threadIdx.x; \
                       i < (n); i += (long)gridDim.x * blockDim.x)

__device__ inline float sigf(float x) { return 1.f / (1.f + expf(-x)); }
__device__ inline float ssqrtf_(float x) { return sqrtf(fmaxf(x, 0.f)) - sqrtf(fmaxf(-x, 0.f)); }

// ---- CDNA5 async global->LDS copy (ASYNCcnt-tracked DMA into LDS) ----------
__device__ inline void async_g2l_b128(int lds_off, const void* gptr) {
  unsigned long long ga = (unsigned long long)gptr;
  asm volatile("global_load_async_to_lds_b128 %0, %1, off"
               :: "v"(lds_off), "v"(ga)
               : "memory");
}
__device__ inline void wait_async0() {
#if __has_builtin(__builtin_amdgcn_s_wait_asynccnt)
  __builtin_amdgcn_s_wait_asynccnt(0);
#else
  asm volatile("s_wait_asynccnt 0x0" ::: "memory");
#endif
}

// A-fragment: 16x32 bf16: lane m (=lane&15) holds K = kh*16 + [0..15]
// contiguous (kh = lane>>4). Two 16-byte runs.
__device__ inline v16bf load_frag_a(const bf16* __restrict__ p) {
  v8bf lo = *(const v8bf*)p;
  v8bf hi = *(const v8bf*)(p + 8);
  return __builtin_shufflevector(lo, hi, 0,1,2,3,4,5,6,7,8,9,10,11,12,13,14,15);
}
// B-fragment: 32x16 bf16: lane n holds column n, K = kh*16 + [0..15] contiguous.
__device__ inline v16bf load_frag_b(const bf16* __restrict__ p) {
  v8bf lo = *(const v8bf*)p;
  v8bf hi = *(const v8bf*)(p + 8);
  return __builtin_shufflevector(lo, hi, 0,1,2,3,4,5,6,7,8,9,10,11,12,13,14,15);
}

// ---------------------------------------------------------------------------
// LDS-staged batched NT GEMM: D[b][M,N] = A[b][M,K] * B[b][N,K]^T
// 64x64 block tile, 8 waves (2x16x16 acc each), K-step 32.
// Double-buffered 2 x (4KB A + 4KB B) LDS, filled with async-to-LDS b128 DMA:
// each of the 256 threads owns one 16B segment of the A tile and one of B.
// K must be a multiple of 32 (buffers zero-padded).
// ---------------------------------------------------------------------------
__global__ __launch_bounds__(256)
void k_wmma_gemm_lds(const bf16* __restrict__ Am, long lda, long sA,
                     const bf16* __restrict__ Bm, long ldb, long sB,
                     float* __restrict__ Dm, long ldd, long sD,
                     int M, int N, int K,
                     const float* __restrict__ biasN, const float* __restrict__ biasM,
                     int act, int accum) {
  extern __shared__ __align__(16) char smem[];   // 16384 bytes
  const long zb = blockIdx.z;
  Am += zb * sA; Bm += zb * sB; Dm += zb * sD;
  const int tid  = threadIdx.x;
  const int lane = tid & 31;
  const int wid  = tid >> 5;
  const int row0 = blockIdx.y * 64;
  const int col0 = blockIdx.x * 64;

  // global->LDS loader mapping (rows clamped to keep addresses valid)
  const int lrow = tid >> 2, lseg = tid & 3;
  int garow = row0 + lrow; if (garow >= M) garow = M - 1;
  int gbrow = col0 + lrow; if (gbrow >= N) gbrow = N - 1;
  const bf16* gA = Am + (long)garow * lda + lseg * 8;
  const bf16* gB = Bm + (long)gbrow * ldb + lseg * 8;
  const int aoff = lrow * 64 + lseg * 16;
  const int boff = 4096 + lrow * 64 + lseg * 16;

  // per-wave compute coordinates
  const int wrow = (wid >> 1) * 16;
  const int wcol = (wid & 1) * 32;
  const int lr = lane & 15, kh = lane >> 4;
  const int saoff  = (wrow + lr) * 64 + kh * 32;
  const int sb0off = 4096 + (wcol + lr) * 64 + kh * 32;
  const int sb1off = 4096 + (wcol + 16 + lr) * 64 + kh * 32;

  // prefetch first K-tile into buffer 0
  async_g2l_b128(aoff, gA);
  async_g2l_b128(boff, gB);

  v8f acc0 = {0.f,0.f,0.f,0.f,0.f,0.f,0.f,0.f};
  v8f acc1 = acc0;
  int buf = 0;
  for (int k = 0; k < K; k += 32) {
    wait_async0();        // my async stores to LDS landed
    __syncthreads();      // everyone's landed; previous buffer fully consumed
    const int cb = buf * 8192;
    if (k + 32 < K) {     // prefetch next K-tile into the other buffer
      const int nb = (buf ^ 1) * 8192;
      async_g2l_b128(nb + aoff, gA + (k + 32));
      async_g2l_b128(nb + boff, gB + (k + 32));
    }
    v8bf a0  = *(const v8bf*)(smem + cb + saoff);
    v8bf a1  = *(const v8bf*)(smem + cb + saoff + 16);
    v8bf b00 = *(const v8bf*)(smem + cb + sb0off);
    v8bf b01 = *(const v8bf*)(smem + cb + sb0off + 16);
    v8bf b10 = *(const v8bf*)(smem + cb + sb1off);
    v8bf b11 = *(const v8bf*)(smem + cb + sb1off + 16);
    v16bf a  = __builtin_shufflevector(a0,  a1,  0,1,2,3,4,5,6,7,8,9,10,11,12,13,14,15);
    v16bf b0 = __builtin_shufflevector(b00, b01, 0,1,2,3,4,5,6,7,8,9,10,11,12,13,14,15);
    v16bf b1 = __builtin_shufflevector(b10, b11, 0,1,2,3,4,5,6,7,8,9,10,11,12,13,14,15);
    acc0 = __builtin_amdgcn_wmma_f32_16x16x32_bf16(false, a, false, b0, (short)0, acc0, false, false);
    acc1 = __builtin_amdgcn_wmma_f32_16x16x32_bf16(false, a, false, b1, (short)0, acc1, false, false);
    buf ^= 1;
  }

  // C/D layout: element r -> M = r + 8*(lane>=16), N = lane&15
  const int mbase = row0 + wrow + (lane >> 4) * 8;
  const int n0 = col0 + wcol + (lane & 15);
  #pragma unroll
  for (int r = 0; r < 8; ++r) {
    int m = mbase + r;
    if (m >= M) continue;
    float bm = biasM ? biasM[m] : 0.f;
    if (n0 < N) {
      float v = acc0[r] + bm + (biasN ? biasN[n0] : 0.f);
      float* pp = Dm + (long)m * ldd + n0;
      if (accum) v += *pp;
      if (act) v = fmaxf(v, 0.f);
      *pp = v;
    }
    int n1 = n0 + 16;
    if (n1 < N) {
      float v = acc1[r] + bm + (biasN ? biasN[n1] : 0.f);
      float* pp = Dm + (long)m * ldd + n1;
      if (accum) v += *pp;
      if (act) v = fmaxf(v, 0.f);
      *pp = v;
    }
  }
}

// ---------------------------------------------------------------------------
// Direct (global->VGPR) NT GEMM, used for skinny tail GEMMs.
// ---------------------------------------------------------------------------
__global__ __launch_bounds__(256)
void k_wmma_gemm_nt(const bf16* __restrict__ Am, long lda, long sA,
                    const bf16* __restrict__ Bm, long ldb, long sB,
                    float* __restrict__ Dm, long ldd, long sD,
                    int M, int N, int K,
                    const float* __restrict__ biasN, const float* __restrict__ biasM,
                    int act, int accum) {
  const long zb = blockIdx.z;
  Am += zb * sA; Bm += zb * sB; Dm += zb * sD;
  const int lane = threadIdx.x & 31;
  const int wid  = threadIdx.x >> 5;
  const int row0 = blockIdx.y * 64 + (wid >> 1) * 16;
  const int col0 = blockIdx.x * 64 + (wid & 1) * 32;
  const int lr   = lane & 15;
  const int kh   = lane >> 4;

  int ar  = row0 + lr;       if (ar  >= M) ar  = M - 1;
  int br0 = col0 + lr;       if (br0 >= N) br0 = N - 1;
  int br1 = col0 + 16 + lr;  if (br1 >= N) br1 = N - 1;
  const bf16* ap  = Am + (long)ar  * lda + kh * 16;
  const bf16* bp0 = Bm + (long)br0 * ldb + kh * 16;
  const bf16* bp1 = Bm + (long)br1 * ldb + kh * 16;

  v8f acc0 = {0.f,0.f,0.f,0.f,0.f,0.f,0.f,0.f};
  v8f acc1 = acc0;
  for (int k = 0; k < K; k += 32) {
    v16bf a  = load_frag_a(ap  + k);
    v16bf b0 = load_frag_b(bp0 + k);
    v16bf b1 = load_frag_b(bp1 + k);
    acc0 = __builtin_amdgcn_wmma_f32_16x16x32_bf16(false, a, false, b0, (short)0, acc0, false, false);
    acc1 = __builtin_amdgcn_wmma_f32_16x16x32_bf16(false, a, false, b1, (short)0, acc1, false, false);
  }
  const int mbase = row0 + (lane >> 4) * 8;
  const int n0 = col0 + (lane & 15);
  #pragma unroll
  for (int r = 0; r < 8; ++r) {
    int m = mbase + r;
    if (m >= M) continue;
    float bm = biasM ? biasM[m] : 0.f;
    if (n0 < N) {
      float v = acc0[r] + bm + (biasN ? biasN[n0] : 0.f);
      float* pp = Dm + (long)m * ldd + n0;
      if (accum) v += *pp;
      if (act) v = fmaxf(v, 0.f);
      *pp = v;
    }
    int n1 = n0 + 16;
    if (n1 < N) {
      float v = acc1[r] + bm + (biasN ? biasN[n1] : 0.f);
      float* pp = Dm + (long)m * ldd + n1;
      if (accum) v += *pp;
      if (act) v = fmaxf(v, 0.f);
      *pp = v;
    }
  }
}

// ---------------------------------------------------------------------------
// Small / fused kernels
// ---------------------------------------------------------------------------
__global__ void k_zero_f32(float* p, long n) { GSL(i, n) p[i] = 0.f; }
__global__ void k_zero_b16(bf16* p, long n)  { GSL(i, n) p[i] = (bf16)0.f; }

__global__ void k_cvt_pad(const float* __restrict__ s, bf16* __restrict__ d,
                          long R, long Cin, long Cpad) {
  GSL(i, R * Cpad) {
    long r = i / Cpad, c = i % Cpad;
    d[i] = (bf16)((c < Cin) ? s[r * Cin + c] : 0.f);
  }
}

// src [batch, R, Nn] f32 -> dst [batch, Nn, Rpad] bf16 (transpose last two, zero pad)
__global__ void k_transpose_pad(const float* __restrict__ s, bf16* __restrict__ d,
                                long batch, long R, long Nn, long Rpad) {
  GSL(i, batch * Nn * Rpad) {
    long r = i % Rpad, n = (i / Rpad) % Nn, b = i / (Rpad * Nn);
    d[i] = (bf16)((r < R) ? s[(b * R + r) * Nn + n] : 0.f);
  }
}

__global__ void k_add2(const float* a, const float* b, float* o, long n) {
  GSL(i, n) o[i] = a[i] + b[i];
}

// x[t*64+b][640] = concat(tanh(emb[data[b,t]]), glove[b,t], zeros)  (bf16)
__global__ void k_embed(const int* __restrict__ data, const float* __restrict__ embw,
                        const float* __restrict__ glove, bf16* __restrict__ x) {
  GSL(i, 1280L * 640) {
    long row = i / 640; int col = (int)(i % 640);
    int t = (int)(row / 64), b = (int)(row % 64);
    float v = 0.f;
    if (col < 300)      { int tk = data[b * 20 + t]; v = tanhf(embw[(long)tk * 300 + col]); }
    else if (col < 600) { v = glove[((long)b * 20 + t) * 300 + (col - 300)]; }
    x[i] = (bf16)v;
  }
}

// gates G[64,4096] (i,f,g,o) -> update c, h(bf16), store h into hall slice
__global__ void k_lstm_cell(const float* __restrict__ G, float* __restrict__ c,
                            bf16* __restrict__ h, bf16* __restrict__ hall_t) {
  GSL(i, 64L * 1024) {
    long bb = i >> 10; int hh = (int)(i & 1023);
    const float* g = G + bb * 4096;
    float gi = g[hh], gf = g[1024 + hh], gg = g[2048 + hh], go = g[3072 + hh];
    float cv = sigf(gf) * c[i] + sigf(gi) * tanhf(gg);
    float hv = sigf(go) * tanhf(cv);
    c[i] = cv;
    bf16 hb = (bf16)hv;
    h[i] = hb; hall_t[i] = hb;
  }
}

// qlogit[b*20+t] = QA1[row=t*64+b,:512] . qa2_w + qa2_b  (one wave per row)
__global__ void k_qlogit(const float* __restrict__ QA1, const float* __restrict__ w,
                         const float* __restrict__ b0, float* __restrict__ out) {
  int row = blockIdx.x * 8 + (threadIdx.x >> 5);
  int lane = threadIdx.x & 31;
  if (row >= 1280) return;
  const float* p = QA1 + (long)row * 512;
  float s = 0.f;
  for (int j = lane; j < 512; j += 32) s += p[j] * w[j];
  for (int o = 16; o; o >>= 1) s += __shfl_xor(s, o);
  if (lane == 0) { int t = row / 64, bb = row % 64; out[bb * 20 + t] = s + b0[0]; }
}

// in-place row softmax, one wave per row
__global__ void k_softmax_rows(float* __restrict__ x, int R, int Nc) {
  int row = blockIdx.x * 8 + (threadIdx.x >> 5);
  int lane = threadIdx.x & 31;
  if (row >= R) return;
  float* p = x + (long)row * Nc;
  float mx = -3.4e38f;
  for (int j = lane; j < Nc; j += 32) mx = fmaxf(mx, p[j]);
  for (int o = 16; o; o >>= 1) mx = fmaxf(mx, __shfl_xor(mx, o));
  float s = 0.f;
  for (int j = lane; j < Nc; j += 32) s += expf(p[j] - mx);
  for (int o = 16; o; o >>= 1) s += __shfl_xor(s, o);
  float inv = 1.f / s;
  for (int j = lane; j < Nc; j += 32) p[j] = expf(p[j] - mx) * inv;
}

__global__ void k_qfeat(const float* __restrict__ attn, const bf16* __restrict__ hall,
                        float* __restrict__ qf) {
  GSL(i, 64L * 1024) {
    long b = i >> 10; int hh = (int)(i & 1023);
    float s = 0.f;
    for (int t = 0; t < 20; ++t)
      s += attn[b * 20 + t] * (float)hall[((long)t * 64 + b) * 1024 + hh];
    qf[i] = s;
  }
}

// pooled weights: w2[b,o,c] = sum_kk qp[b,o*5+kk] * W[o*5+kk, c]   (bf16 out, zero pad)
__global__ void k_pool_w2(const float* __restrict__ qp, const float* __restrict__ W,
                          bf16* __restrict__ w2, long Cin, long Cpad) {
  GSL(i, 64L * 1000 * Cpad) {
    long c = i % Cpad, o = (i / Cpad) % 1000, b = i / (Cpad * 1000);
    float s = 0.f;
    if (c < Cin) {
      #pragma unroll
      for (int kk = 0; kk < 5; ++kk)
        s += qp[b * 5000 + o * 5 + kk] * W[(o * 5 + kk) * Cin + c];
    }
    w2[i] = (bf16)s;
  }
}

__global__ void k_pool_zb(const float* __restrict__ qp, const float* __restrict__ bias,
                          float* __restrict__ zb) {
  GSL(i, 64L * 1000) {
    long b = i / 1000, o = i % 1000;
    float s = 0.f;
    #pragma unroll
    for (int kk = 0; kk < 5; ++kk) s += qp[b * 5000 + o * 5 + kk] * bias[o * 5 + kk];
    zb[i] = s;
  }
}

// z += zb ; signed sqrt ; accumulate sum-of-squares per batch (wave reduce + atomic)
// per-batch region (Onum*Nn) is a multiple of 32 for 1000*196 (each wave in one b), not for 1000*50.
__global__ void k_ssqrt_norm(float* __restrict__ z, const float* __restrict__ zb,
                             float* __restrict__ nrm, long Onum, long Nn) {
  long perb = Onum * Nn;
  GSL(i, 64L * perb) {
    long b = i / perb, o = (i % perb) / Nn;
    float v = ssqrtf_(z[i] + zb[b * Onum + o]);
    z[i] = v;
    float sq = v * v;
    #pragma unroll
    for (int ofs = 16; ofs; ofs >>= 1) sq += __shfl_xor(sq, ofs);
    if ((threadIdx.x & 31) == 0) atomicAdd(&nrm[b], sq);
  }
}

__global__ void k_scale(float* __restrict__ z, const float* __restrict__ nrm, long perb) {
  GSL(i, 64L * perb) {
    long b = i / perb;
    z[i] /= fmaxf(sqrtf(nrm[b]), 1e-12f);
  }
}

// logits[b,g,n] = a2w[g,:] . att1[b,:,n] + a2b[g]
__global__ void k_att_logit(const float* __restrict__ att1, const float* __restrict__ a2w,
                            const float* __restrict__ a2b, float* __restrict__ out,
                            long G, long Nn) {
  GSL(i, 64L * G * Nn) {
    long n = i % Nn, g = (i / Nn) % G, b = i / (Nn * G);
    float s = a2b[g];
    for (int ss = 0; ss < 512; ++ss)
      s += a2w[g * 512 + ss] * att1[(b * 512 + ss) * Nn + n];
    out[i] = s;
  }
}

// out[b, g*Cc + c] = sum_n attn[b,g,n] * feat(b,c,n) with generic strides
__global__ void k_att_apply(const float* __restrict__ attn, const float* __restrict__ feat,
                            float* __restrict__ out, long G, long Cc, long Nn,
                            long bs, long cs, long ns) {
  GSL(i, 64L * G * Cc) {
    long c = i % Cc, g = (i / Cc) % G, b = i / (Cc * G);
    float s = 0.f;
    const float* ar = attn + (b * G + g) * Nn;
    for (long n = 0; n < Nn; ++n) s += ar[n] * feat[b * bs + c * cs + n * ns];
    out[i] = s;
  }
}

// MFB fuse + signed sqrt + L2 normalize -> shared[b*2000 + o] (LDS staging + reduce)
__global__ __launch_bounds__(256)
void k_fuse(const float* __restrict__ u, const float* __restrict__ v, float* __restrict__ out) {
  __shared__ float zf[1000];
  __shared__ float red[256];
  int b = blockIdx.x, tid = threadIdx.x;
  const float* uu = u + (long)b * 5000;
  const float* vv = v + (long)b * 5000;
  float loc = 0.f;
  for (int o = tid; o < 1000; o += 256) {
    float s = 0.f;
    #pragma unroll
    for (int kk = 0; kk < 5; ++kk) s += uu[o * 5 + kk] * vv[o * 5 + kk];
    float ss = ssqrtf_(s);
    zf[o] = ss; loc += ss * ss;
  }
  red[tid] = loc; __syncthreads();
  for (int st = 128; st; st >>= 1) { if (tid < st) red[tid] += red[tid + st]; __syncthreads(); }
  float nrm = fmaxf(sqrtf(red[0]), 1e-12f);
  for (int o = tid; o < 1000; o += 256) out[(long)b * 2000 + o] = zf[o] / nrm;
}

__global__ __launch_bounds__(256)
void k_logsoftmax(const float* __restrict__ x, float* __restrict__ out) {
  __shared__ float red[256];
  int b = blockIdx.x, tid = threadIdx.x;
  const float* p = x + (long)b * 3000;
  float mx = -3.4e38f;
  for (int j = tid; j < 3000; j += 256) mx = fmaxf(mx, p[j]);
  red[tid] = mx; __syncthreads();
  for (int st = 128; st; st >>= 1) { if (tid < st) red[tid] = fmaxf(red[tid], red[tid + st]); __syncthreads(); }
  mx = red[0]; __syncthreads();
  float s = 0.f;
  for (int j = tid; j < 3000; j += 256) s += expf(p[j] - mx);
  red[tid] = s; __syncthreads();
  for (int st = 128; st; st >>= 1) { if (tid < st) red[tid] += red[tid + st]; __syncthreads(); }
  float lse = logf(red[0]);
  for (int j = tid; j < 3000; j += 256) out[(long)b * 3000 + j] = p[j] - mx - lse;
}

__global__ void k_binary(const float* __restrict__ sh, const float* __restrict__ bw,
                         const float* __restrict__ bb, float* __restrict__ out) {
  int b = blockIdx.x, lane = threadIdx.x;
  float s = 0.f;
  for (int j = lane; j < 2000; j += 32) s += sh[(long)b * 2000 + j] * bw[j];
  for (int o = 16; o; o >>= 1) s += __shfl_xor(s, o);
  if (lane == 0) out[b] = 1.f / (1.f + expf(-(s + bb[0])));
}

// ---------------------------------------------------------------------------
// Host driver
// ---------------------------------------------------------------------------
extern "C" void kernel_launch(void* const* d_in, const int* in_sizes, int n_in,
                              void* d_out, int out_size, void* d_ws, size_t ws_size,
                              hipStream_t stream) {
  (void)in_sizes; (void)n_in; (void)out_size; (void)ws_size;
  const int*   data  = (const int*)d_in[0];
  const float* img   = (const float*)d_in[1];
  const float* glove = (const float*)d_in[2];
  const float* tok   = (const float*)d_in[4];
  const float* embw  = (const float*)d_in[5];
  const float* W_ih  = (const float*)d_in[6];
  const float* W_hh  = (const float*)d_in[7];
  const float* b_ih  = (const float*)d_in[8];
  const float* b_hh  = (const float*)d_in[9];
  const float* q1_w = (const float*)d_in[10]; const float* q1_b = (const float*)d_in[11];
  const float* q2_w = (const float*)d_in[12]; const float* q2_b = (const float*)d_in[13];
  const float* q3_w = (const float*)d_in[14]; const float* q3_b = (const float*)d_in[15];
  const float* q4_w = (const float*)d_in[16]; const float* q4_b = (const float*)d_in[17];
  const float* ci_w = (const float*)d_in[18]; const float* ci_b = (const float*)d_in[19];
  const float* co_w = (const float*)d_in[20]; const float* co_b = (const float*)d_in[21];
  const float* qa1_w = (const float*)d_in[22]; const float* qa1_b = (const float*)d_in[23];
  const float* qa2_w = (const float*)d_in[24]; const float* qa2_b = (const float*)d_in[25];
  const float* ia1_w = (const float*)d_in[26]; const float* ia1_b = (const float*)d_in[27];
  const float* ia2_w = (const float*)d_in[28]; const float* ia2_b = (const float*)d_in[29];
  const float* oa1_w = (const float*)d_in[30]; const float* oa1_b = (const float*)d_in[31];
  const float* oa2_w = (const float*)d_in[32]; const float* oa2_b = (const float*)d_in[33];
  const float* li_w = (const float*)d_in[34]; const float* li_b = (const float*)d_in[35];
  const float* lo_w = (const float*)d_in[36]; const float* lo_b = (const float*)d_in[37];
  const float* pred_w = (const float*)d_in[38]; const float* pred_b = (const float*)d_in[39];
  const float* bin_w = (const float*)d_in[40]; const float* bin_b = (const float*)d_in[41];
  float* out = (float*)d_out;

  char* wsb = (char*)d_ws;
  size_t cur = 0;
  auto alloc = [&](size_t bytes) -> void* {
    void* p = wsb + cur;
    cur += (bytes + 255) & ~(size_t)255;
    return p;
  };
  bf16*  xbf    = (bf16*) alloc(1280L * 640 * 2);
  bf16*  wihb   = (bf16*) alloc(4096L * 640 * 2);
  bf16*  whhb   = (bf16*) alloc(4096L * 1024 * 2);
  float* bsum   = (float*)alloc(4096L * 4);
  float* gates  = (float*)alloc(1280L * 4096 * 4);
  bf16*  hbf    = (bf16*) alloc(64L * 1024 * 2);
  float* cst    = (float*)alloc(64L * 1024 * 4);
  bf16*  hallb  = (bf16*) alloc(1280L * 1024 * 2);
  bf16*  qa1wb  = (bf16*) alloc(512L * 1024 * 2);
  float* QA1    = (float*)alloc(1280L * 512 * 4);
  float* qlog   = (float*)alloc(64L * 20 * 4);
  float* qfeat  = (float*)alloc(64L * 1024 * 4);
  bf16*  qfeatb = (bf16*) alloc(64L * 1024 * 2);
  bf16*  q1wb   = (bf16*) alloc(5000L * 1024 * 2);
  bf16*  q2wb   = (bf16*) alloc(5000L * 1024 * 2);
  bf16*  q3wb   = (bf16*) alloc(5000L * 1024 * 2);
  bf16*  q4wb   = (bf16*) alloc(5000L * 1024 * 2);
  float* qp1    = (float*)alloc(64L * 5000 * 4);
  float* qp2    = (float*)alloc(64L * 5000 * 4);
  float* qp3    = (float*)alloc(64L * 5000 * 4);
  float* qp4    = (float*)alloc(64L * 5000 * 4);
  bf16*  imgTb  = (bf16*) alloc(64L * 196 * 2048 * 2);
  bf16*  w2i    = (bf16*) alloc(64L * 1000 * 2048 * 2);
  float* zbi    = (float*)alloc(64L * 1000 * 4);
  float* zi     = (float*)alloc(64L * 1000 * 196 * 4);
  float* nrmi   = (float*)alloc(64L * 4);
  bf16*  zTib   = (bf16*) alloc(64L * 196 * 1024 * 2);
  bf16*  ia1wb  = (bf16*) alloc(512L * 1024 * 2);
  float* att1i  = (float*)alloc(64L * 512 * 196 * 4);
  float* ilog   = (float*)alloc(64L * 2 * 196 * 4);
  float* iatt   = (float*)alloc(64L * 4096 * 4);
  bf16*  iattb  = (bf16*) alloc(64L * 4096 * 2);
  bf16*  liwb   = (bf16*) alloc(5000L * 4096 * 2);
  float* vi     = (float*)alloc(64L * 5000 * 4);
  bf16*  tokb   = (bf16*) alloc(3200L * 320 * 2);
  bf16*  w2o    = (bf16*) alloc(64L * 1000 * 320 * 2);
  float* zbo    = (float*)alloc(64L * 1000 * 4);
  float* zo     = (float*)alloc(64L * 1000 * 50 * 4);
  float* nrmo   = (float*)alloc(64L * 4);
  bf16*  zTob   = (bf16*) alloc(64L * 50 * 1024 * 2);
  bf16*  oa1wb  = (bf16*) alloc(512L * 1024 * 2);
  float* att1o  = (float*)alloc(64L * 512 * 50 * 4);
  float* olog   = (float*)alloc(64L * 2 * 50 * 4);
  float* oatt   = (float*)alloc(64L * 600 * 4);
  bf16*  oattb  = (bf16*) alloc(64L * 640 * 2);
  bf16*  lowb   = (bf16*) alloc(5000L * 640 * 2);
  float* vo     = (float*)alloc(64L * 5000 * 4);
  float* sharedf= (float*)alloc(64L * 2000 * 4);
  bf16*  sharedb= (bf16*) alloc(64L * 2016 * 2);
  bf16*  predwb = (bf16*) alloc(3000L * 2016 * 2);
  float* plin   = (float*)alloc(64L * 3000 * 4);

  auto gs = [](long n) { long b = (n + 255) / 256; if (b > 16384) b = 16384; return dim3((unsigned)b); };
  // LDS-staged async GEMM (preferred path)
  auto gemmL = [&](const bf16* Ap, long lda, long sA, const bf16* Bp, long ldb, long sB,
                   float* Dp, long ldd, long sD, int M, int N, int K,
                   const float* bN, const float* bM, int act, int accum, int batch) {
    dim3 grid((unsigned)((N + 63) / 64), (unsigned)((M + 63) / 64), (unsigned)batch);
    k_wmma_gemm_lds<<<grid, 256, 16384, stream>>>(Ap, lda, sA, Bp, ldb, sB, Dp, ldd, sD,
                                                  M, N, K, bN, bM, act, accum);
  };
  // direct-load GEMM (skinny tails)
  auto gemmD = [&](const bf16* Ap, long lda, long sA, const bf16* Bp, long ldb, long sB,
                   float* Dp, long ldd, long sD, int M, int N, int K,
                   const float* bN, const float* bM, int act, int accum, int batch) {
    dim3 grid((unsigned)((N + 63) / 64), (unsigned)((M + 63) / 64), (unsigned)batch);
    k_wmma_gemm_nt<<<grid, 256, 0, stream>>>(Ap, lda, sA, Bp, ldb, sB, Dp, ldd, sD,
                                             M, N, K, bN, bM, act, accum);
  };

  // ---- bf16 conversions (zero-padded K to multiples of 32) ----
  k_cvt_pad<<<gs(4096L*640),  256, 0, stream>>>(W_ih,  wihb,  4096, 600, 640);
  k_cvt_pad<<<gs(4096L*1024), 256, 0, stream>>>(W_hh,  whhb,  4096, 1024, 1024);
  k_cvt_pad<<<gs(512L*1024),  256, 0, stream>>>(qa1_w, qa1wb, 512, 1024, 1024);
  k_cvt_pad<<<gs(5000L*1024), 256, 0, stream>>>(q1_w,  q1wb,  5000, 1024, 1024);
  k_cvt_pad<<<gs(5000L*1024), 256, 0, stream>>>(q2_w,  q2wb,  5000, 1024, 1024);
  k_cvt_pad<<<gs(5000L*1024), 256, 0, stream>>>(q3_w,  q3wb,  5000, 1024, 1024);
  k_cvt_pad<<<gs(5000L*1024), 256, 0, stream>>>(q4_w,  q4wb,  5000, 1024, 1024);
  k_cvt_pad<<<gs(512L*1024),  256, 0, stream>>>(ia1_w, ia1wb, 512, 1000, 1024);
  k_cvt_pad<<<gs(512L*1024),  256, 0, stream>>>(oa1_w, oa1wb, 512, 1000, 1024);
  k_cvt_pad<<<gs(5000L*4096), 256, 0, stream>>>(li_w,  liwb,  5000, 4096, 4096);
  k_cvt_pad<<<gs(5000L*640),  256, 0, stream>>>(lo_w,  lowb,  5000, 600, 640);
  k_cvt_pad<<<gs(3200L*320),  256, 0, stream>>>(tok,   tokb,  3200, 300, 320);
  k_cvt_pad<<<gs(3000L*2016), 256, 0, stream>>>(pred_w, predwb, 3000, 2000, 2016);
  k_transpose_pad<<<gs(64L*196*2048), 256, 0, stream>>>(img, imgTb, 64, 2048, 196, 2048);
  k_add2<<<gs(4096), 256, 0, stream>>>(b_ih, b_hh, bsum, 4096);
  k_embed<<<gs(1280L*640), 256, 0, stream>>>(data, embw, glove, xbf);
  k_zero_f32<<<gs(64L*1024), 256, 0, stream>>>(cst, 64L*1024);
  k_zero_b16<<<gs(64L*1024), 256, 0, stream>>>(hbf, 64L*1024);
  k_zero_f32<<<1, 256, 0, stream>>>(nrmi, 64);
  k_zero_f32<<<1, 256, 0, stream>>>(nrmo, 64);

  // ---- LSTM: batch input projection, then 20 recurrent steps ----
  gemmL(xbf, 640, 0, wihb, 640, 0, gates, 4096, 0, 1280, 4096, 640, bsum, nullptr, 0, 0, 1);
  for (int t = 0; t < 20; ++t) {
    gemmL(hbf, 1024, 0, whhb, 1024, 0, gates + (long)t*64*4096, 4096, 0,
          64, 4096, 1024, nullptr, nullptr, 0, 1, 1);
    k_lstm_cell<<<gs(64L*1024), 256, 0, stream>>>(gates + (long)t*64*4096, cst, hbf,
                                                  hallb + (long)t*64*1024);
  }

  // ---- question attention -> q_feat ----
  gemmL(hallb, 1024, 0, qa1wb, 1024, 0, QA1, 512, 0, 1280, 512, 1024, qa1_b, nullptr, 1, 0, 1);
  k_qlogit<<<dim3(160), 256, 0, stream>>>(QA1, qa2_w, qa2_b, qlog);
  k_softmax_rows<<<dim3(8), 256, 0, stream>>>(qlog, 64, 20);
  k_qfeat<<<gs(64L*1024), 256, 0, stream>>>(qlog, hallb, qfeat);
  k_cvt_pad<<<gs(64L*1024), 256, 0, stream>>>(qfeat, qfeatb, 64, 1024, 1024);

  // ---- four question projections ----
  gemmL(qfeatb, 1024, 0, q1wb, 1024, 0, qp1, 5000, 0, 64, 5000, 1024, q1_b, nullptr, 0, 0, 1);
  gemmL(qfeatb, 1024, 0, q2wb, 1024, 0, qp2, 5000, 0, 64, 5000, 1024, q2_b, nullptr, 0, 0, 1);
  gemmL(qfeatb, 1024, 0, q3wb, 1024, 0, qp3, 5000, 0, 64, 5000, 1024, q3_b, nullptr, 0, 0, 1);
  gemmL(qfeatb, 1024, 0, q4wb, 1024, 0, qp4, 5000, 0, 64, 5000, 1024, q4_b, nullptr, 0, 0, 1);

  // ---- image branch (weight-pooled MFB attention) ----
  k_pool_w2<<<gs(64L*1000*2048), 256, 0, stream>>>(qp1, ci_w, w2i, 2048, 2048);
  k_pool_zb<<<gs(64L*1000), 256, 0, stream>>>(qp1, ci_b, zbi);
  gemmL(w2i, 2048, 1000L*2048, imgTb, 2048, 196L*2048, zi, 196, 1000L*196,
        1000, 196, 2048, nullptr, nullptr, 0, 0, 64);
  k_ssqrt_norm<<<gs(64L*196000), 256, 0, stream>>>(zi, zbi, nrmi, 1000, 196);
  k_scale<<<gs(64L*196000), 256, 0, stream>>>(zi, nrmi, 196000);
  k_transpose_pad<<<gs(64L*196*1024), 256, 0, stream>>>(zi, zTib, 64, 1000, 196, 1024);
  gemmL(ia1wb, 1024, 0, zTib, 1024, 196L*1024, att1i, 196, 512L*196,
        512, 196, 1024, nullptr, ia1_b, 1, 0, 64);
  k_att_logit<<<gs(64L*2*196), 256, 0, stream>>>(att1i, ia2_w, ia2_b, ilog, 2, 196);
  k_softmax_rows<<<dim3(16), 256, 0, stream>>>(ilog, 128, 196);
  k_att_apply<<<gs(64L*2*2048), 256, 0, stream>>>(ilog, img, iatt, 2, 2048, 196,
                                                  (long)2048*196, 196, 1);
  k_cvt_pad<<<gs(64L*4096), 256, 0, stream>>>(iatt, iattb, 64, 4096, 4096);
  gemmL(iattb, 4096, 0, liwb, 4096, 0, vi, 5000, 0, 64, 5000, 4096, li_b, nullptr, 0, 0, 1);
  k_fuse<<<dim3(64), 256, 0, stream>>>(qp2, vi, sharedf + 0);

  // ---- OCR branch ----
  k_pool_w2<<<gs(64L*1000*320), 256, 0, stream>>>(qp3, co_w, w2o, 300, 320);
  k_pool_zb<<<gs(64L*1000), 256, 0, stream>>>(qp3, co_b, zbo);
  gemmL(w2o, 320, 1000L*320, tokb, 320, 50L*320, zo, 50, 1000L*50,
        1000, 50, 320, nullptr, nullptr, 0, 0, 64);
  k_ssqrt_norm<<<gs(64L*50000), 256, 0, stream>>>(zo, zbo, nrmo, 1000, 50);
  k_scale<<<gs(64L*50000), 256, 0, stream>>>(zo, nrmo, 50000);
  k_transpose_pad<<<gs(64L*50*1024), 256, 0, stream>>>(zo, zTob, 64, 1000, 50, 1024);
  gemmL(oa1wb, 1024, 0, zTob, 1024, 50L*1024, att1o, 50, 512L*50,
        512, 50, 1024, nullptr, oa1_b, 1, 0, 64);
  k_att_logit<<<gs(64L*2*50), 256, 0, stream>>>(att1o, oa2_w, oa2_b, olog, 2, 50);
  k_softmax_rows<<<dim3(16), 256, 0, stream>>>(olog, 128, 50);
  k_att_apply<<<gs(64L*2*300), 256, 0, stream>>>(olog, tok, oatt, 2, 300, 50,
                                                 (long)50*300, 1, 300);
  k_cvt_pad<<<gs(64L*640), 256, 0, stream>>>(oatt, oattb, 64, 600, 640);
  gemmD(oattb, 640, 0, lowb, 640, 0, vo, 5000, 0, 64, 5000, 640, lo_b, nullptr, 0, 0, 1);
  k_fuse<<<dim3(64), 256, 0, stream>>>(qp4, vo, sharedf + 1000);

  // ---- heads ----
  k_cvt_pad<<<gs(64L*2016), 256, 0, stream>>>(sharedf, sharedb, 64, 2000, 2016);
  gemmD(sharedb, 2016, 0, predwb, 2016, 0, plin, 3000, 0, 64, 3000, 2016, pred_b, nullptr, 0, 0, 1);
  k_logsoftmax<<<dim3(64), 256, 0, stream>>>(plin, out + 64);
  k_binary<<<dim3(64), 32, 0, stream>>>(sharedf, bin_w, bin_b, out);
}
